// AdaptiveLinearWithChannelHyper_36481452212286
// MI455X (gfx1250) — matrix operand order repaired and checked by reading, not compiled, not run on hardware
//
#include <hip/hip_runtime.h>

// CDNA5 (gfx1250) wave32 bf16x3 split-accumulate batched-GEMM:
//   out[ch] = x[ch] (2048x256) @ weight[indices[ch], t] (256x256) + bias[indices[ch], t]
//
// f32 operands split into bf16 hi (trunc) + bf16 lo (RNE residual);
// product = xh*wh + xh*wl + xl*wh via V_WMMA_F32_16X16X32_BF16.
//
// Weight persistence: full 128(n) x 256(k) W slab staged once per block as
// bf16 hi/lo in LDS (130 KiB of the CDNA5 320 KiB WGP LDS), reused across 4
// M-blocks. hi/lo are interleaved per row so every fragment load is a
// ds_load_b128 with a 16-bit immediate offset from one per-lane base ->
// no in-loop VALU address math, no WMMA->VALU WAR hazard nops.

typedef __bf16 v16bf __attribute__((ext_vector_type(16)));
typedef __bf16 v8bf  __attribute__((ext_vector_type(8)));
typedef float  v8f   __attribute__((ext_vector_type(8)));

#define N_TILES 8
#define D_IN    256
#define D_OUT   256
#define PTS     2048
#define KT      64     // X K-step staged per iteration (2 WMMA chunks of 32)
#define MT      128
#define NT      128
#define MGRP    4      // M-blocks processed per workgroup
#define XSTR    136    // X LDS row stride: hi[64] | lo[64] | pad[8]  (272B rows)
#define XLO     64     // lo offset within an X row (elems)
#define WSTR    520    // W LDS row stride: hi[256] | lo[256] | pad[8] (1040B rows)
#define WLO     256    // lo offset within a W row (elems)

static __device__ __forceinline__ unsigned int bf16_rne_bits(float f) {
    unsigned int u = __float_as_uint(f);
    unsigned int r = 0x7FFFu + ((u >> 16) & 1u);
    return (u + r) >> 16;
}

// hi = truncated bf16(f); lo = rne bf16(f - hi)   (16 effective mantissa bits)
static __device__ __forceinline__ void bf16_split(float f, unsigned int& hi, unsigned int& lo) {
    unsigned int u = __float_as_uint(f);
    hi = u >> 16;
    float hf = __uint_as_float(u & 0xFFFF0000u);
    lo = bf16_rne_bits(f - hf);
}

static __device__ __forceinline__ v16bf join16(v8bf a, v8bf b) {
    return __builtin_shufflevector(a, b, 0, 1, 2, 3, 4, 5, 6, 7,
                                         8, 9, 10, 11, 12, 13, 14, 15);
}

__global__ __launch_bounds__(256, 1)
void alwc_wmma_bf16x3_persistW(const float* __restrict__ x,
                               const float* __restrict__ weight,
                               const float* __restrict__ bias,
                               const int*   __restrict__ indices,
                               const int*   __restrict__ t_ptr,
                               float*       __restrict__ out)
{
    extern __shared__ unsigned short smem[];
    unsigned short* wt = smem;                 // 128 x 520 bf16 (hi|lo rows)
    unsigned short* xs = wt + NT * WSTR;       // 128 x 136 bf16 (hi|lo rows)

    const int t    = t_ptr[0];
    const int ch   = blockIdx.z;
    const int idx  = indices[ch];
    const int n0   = blockIdx.x * NT;

    const int tid  = threadIdx.x;
    const int lane = tid & 31;
    const int wv   = tid >> 5;
    const int wm   = (wv & 3) * 32;    // wave M offset (4 waves along M)
    const int wn   = (wv >> 2) * 64;   // wave N offset (2 waves along N)
    const int half = lane >> 4;
    const int l16  = lane & 15;

    const float* wg = weight + (((size_t)idx * N_TILES + t) * D_IN) * D_OUT + n0;
    const float* bg = bias   + ((size_t)idx * N_TILES + t) * D_OUT + n0;

    // ---- One-time W stage: full 256(k) x 128(n) slab, transposed to (n,k) ----
    // Thread owns column n = tid&127, 4 K-contiguous values per pass
    // (lane-coalesced b32 global loads, K-packed b64 LDS stores).
    {
        const int wcn = tid & 127;
        const int wkb = (tid >> 7) * 4;    // 0 or 4 (+8 per pass)
#pragma unroll 4
        for (int p = 0; p < D_IN / 8; ++p) {
            const int k = p * 8 + wkb;
            const float* wp = wg + (size_t)k * D_OUT + wcn;
            unsigned int h0, l0, h1, l1, h2, l2, h3, l3;
            bf16_split(wp[0],         h0, l0);
            bf16_split(wp[D_OUT],     h1, l1);
            bf16_split(wp[2 * D_OUT], h2, l2);
            bf16_split(wp[3 * D_OUT], h3, l3);
            const int wo = wcn * WSTR + k;
            *(uint2*)&wt[wo]       = make_uint2(h0 | (h1 << 16), h2 | (h3 << 16));
            *(uint2*)&wt[wo + WLO] = make_uint2(l0 | (l1 << 16), l2 | (l3 << 16));
        }
    }
    __syncthreads();

    // X staging coords: 128 rows x 16 float4 per K-step -> 8 passes
    const int xr = tid >> 4;            // 0..15 (+16 per pass)
    const int xc = (tid & 15) * 4;      // 0..60

    // Per-lane fragment base pointers (half-dependent part folded in once).
    const unsigned short* xsb = xs + (wm + l16) * XSTR + 16 * half;
    const unsigned short* wtb = wt + (wn + l16) * WSTR + 16 * half;

    // ---- Loop over 4 M-blocks reusing the persistent W slab ----
    for (int mb = 0; mb < MGRP; ++mb) {
        const int m0 = (blockIdx.y * MGRP + mb) * MT;
        const float* xg = x + ((size_t)ch * PTS + m0) * D_IN;

        float4 xreg[8];
#pragma unroll
        for (int p = 0; p < 8; ++p)
            xreg[p] = *(const float4*)(xg + (size_t)(xr + p * 16) * D_IN + xc);

        v8f acc[2][4];
#pragma unroll
        for (int ms = 0; ms < 2; ++ms)
#pragma unroll
            for (int ns = 0; ns < 4; ++ns)
                acc[ms][ns] = (v8f)(0.0f);

#pragma unroll 1
        for (int s = 0; s < D_IN / KT; ++s) {
            // Split staged X f32 to bf16 hi/lo and commit to LDS.
#pragma unroll
            for (int p = 0; p < 8; ++p) {
                unsigned int h0, l0, h1, l1, h2, l2, h3, l3;
                float4 v = xreg[p];
                bf16_split(v.x, h0, l0); bf16_split(v.y, h1, l1);
                bf16_split(v.z, h2, l2); bf16_split(v.w, h3, l3);
                const int xo = (xr + p * 16) * XSTR + xc;
                *(uint2*)&xs[xo]       = make_uint2(h0 | (h1 << 16), h2 | (h3 << 16));
                *(uint2*)&xs[xo + XLO] = make_uint2(l0 | (l1 << 16), l2 | (l3 << 16));
            }
            __syncthreads();

            // Prefetch next K step while this one computes.
            if (s + 1 < D_IN / KT) {
                const int k0 = (s + 1) * KT;
#pragma unroll
                for (int p = 0; p < 8; ++p)
                    xreg[p] = *(const float4*)(xg + (size_t)(xr + p * 16) * D_IN + k0 + xc);
            }

            const unsigned short* wts = wtb + s * KT;   // single per-step add

            // 2 K-chunks of 32 -> 48 WMMAs per wave per step.
#pragma unroll
            for (int kc = 0; kc < KT / 32; ++kc) {
                const int kb = kc * 32;
                // A fragments: elems 0-15 = K(16*half .. +15)
                v16bf ah[2], al[2];
#pragma unroll
                for (int ms = 0; ms < 2; ++ms) {
                    const unsigned short* ap = xsb + ms * 16 * XSTR + kb;
                    ah[ms] = join16(*(const v8bf*)(ap),
                                    *(const v8bf*)(ap + 8));
                    al[ms] = join16(*(const v8bf*)(ap + XLO),
                                    *(const v8bf*)(ap + XLO + 8));
                }
#pragma unroll
                for (int ns = 0; ns < 4; ++ns) {
                    // B fragment: elems 0-15 = K(16*half .. +15), N = lane&15.
                    const unsigned short* bp = wts + ns * 16 * WSTR + kb;
                    v16bf bh = join16(*(const v8bf*)(bp),
                                      *(const v8bf*)(bp + 8));
                    v16bf bl = join16(*(const v8bf*)(bp + WLO),
                                      *(const v8bf*)(bp + WLO + 8));
#pragma unroll
                    for (int ms = 0; ms < 2; ++ms) {
                        acc[ms][ns] = __builtin_amdgcn_wmma_f32_16x16x32_bf16(
                            false, ah[ms], false, bh, (short)0, acc[ms][ns], false, false);
                        acc[ms][ns] = __builtin_amdgcn_wmma_f32_16x16x32_bf16(
                            false, ah[ms], false, bl, (short)0, acc[ms][ns], false, false);
                        acc[ms][ns] = __builtin_amdgcn_wmma_f32_16x16x32_bf16(
                            false, al[ms], false, bh, (short)0, acc[ms][ns], false, false);
                    }
                }
            }
            __syncthreads();
        }

        // Epilogue: bias add + store. f32 C/D layout: VGPR v = row v + 8*half,
        // column = lane&15 within the 16x16 subtile.
#pragma unroll
        for (int ns = 0; ns < 4; ++ns) {
            const int ncol = wn + ns * 16 + l16;
            const float bv = bg[ncol];
#pragma unroll
            for (int ms = 0; ms < 2; ++ms) {
                const int rbase = m0 + wm + ms * 16 + 8 * half;
                float* op = out + ((size_t)ch * PTS + rbase) * D_OUT + n0 + ncol;
#pragma unroll
                for (int v = 0; v < 8; ++v)
                    op[(size_t)v * D_OUT] = acc[ms][ns][v] + bv;
            }
        }
    }
}

extern "C" void kernel_launch(void* const* d_in, const int* in_sizes, int n_in,
                              void* d_out, int out_size, void* d_ws, size_t ws_size,
                              hipStream_t stream) {
    const float* x       = (const float*)d_in[0];
    const float* weight  = (const float*)d_in[1];
    const float* bias    = (const float*)d_in[2];
    const int*   indices = (const int*)d_in[3];
    const int*   t_ptr   = (const int*)d_in[4];
    float*       out     = (float*)d_out;

    const int n_sel = in_sizes[3];
    const size_t shmem = (size_t)(NT * WSTR + MT * XSTR) * sizeof(unsigned short);
    dim3 grid(D_OUT / NT, (PTS / MT) / MGRP, n_sel);
    alwc_wmma_bf16x3_persistW<<<grid, 256, shmem, stream>>>(x, weight, bias, indices, t_ptr, out);
}
